// GraphEncoder_62354335203676
// MI455X (gfx1250) — matrix operand order repaired and checked by reading, not compiled, not run on hardware
//
#include <hip/hip_runtime.h>
#include <hip/hip_bf16.h>

typedef __attribute__((ext_vector_type(16))) __bf16 v16bf;
typedef __attribute__((ext_vector_type(8)))  float  v8f;

__device__ __forceinline__ unsigned short f32_to_bf16(float f) {
    unsigned int u = __float_as_uint(f);
    unsigned int r = 0x7FFFu + ((u >> 16) & 1u);   // round-to-nearest-even
    u += r;
    return (unsigned short)(u >> 16);
}

// ---------------- zero fill ----------------
__global__ void zero_kernel(float* __restrict__ p, int n) {
    int i = blockIdx.x * blockDim.x + threadIdx.x;
    int stride = gridDim.x * blockDim.x;
    for (; i < n; i += stride) p[i] = 0.0f;
}

// ---------------- degree + rsqrt ----------------
__global__ void deg_kernel(const long long* __restrict__ dst, float* __restrict__ deg, int E) {
    int e = blockIdx.x * blockDim.x + threadIdx.x;
    if (e < E) atomicAdd(&deg[(int)dst[e]], 1.0f);
}

__global__ void dinv_kernel(const float* __restrict__ deg, float* __restrict__ dinv, int n) {
    int i = blockIdx.x * blockDim.x + threadIdx.x;
    if (i < n) dinv[i] = rsqrtf(deg[i] + 1.0f);
}

// ---------------- pack A (f32 [N,128] -> bf16 WMMA A-layout) ----------------
// Layout: [rowTile][kt=0..3][lane=0..31][elem=0..15], lane L: h=L/16, m=L%16
// elem i -> K = kt*32 + h*16 + i
__global__ void pack_a_kernel(const float* __restrict__ X, unsigned short* __restrict__ Ap, int rowTiles) {
    int t = blockIdx.x * blockDim.x + threadIdx.x;
    int total = rowTiles * 4 * 32;
    if (t >= total) return;
    int lane = t & 31;
    int kt   = (t >> 5) & 3;
    int rt   = t >> 7;
    int h = lane >> 4;
    int m = lane & 15;
    const float* xr = X + (size_t)(rt * 16 + m) * 128 + kt * 32;
    unsigned short* o = Ap + (size_t)t * 16;
#pragma unroll
    for (int i = 0; i < 8; ++i) o[i]     = f32_to_bf16(xr[h * 16 + i]);
#pragma unroll
    for (int i = 0; i < 8; ++i) o[8 + i] = f32_to_bf16(xr[h * 16 + 8 + i]);
}

// ---------------- pack B (f32 [128, NT*16] -> bf16 WMMA B-layout) ----------------
// Layout: [kt=0..3][nt=0..NT-1][lane][elem], lane L: h=L/16, n=nt*16+L%16
// elem i -> K = kt*32 + h*16 + i
__global__ void pack_b_kernel(const float* __restrict__ W, unsigned short* __restrict__ Bp, int NT) {
    int t = blockIdx.x * blockDim.x + threadIdx.x;
    int total = 4 * NT * 32;
    if (t >= total) return;
    int lane = t & 31;
    int nt   = (t >> 5) % NT;
    int kt   = (t >> 5) / NT;
    int h = lane >> 4;
    int n = nt * 16 + (lane & 15);
    int Ncols = NT * 16;
    unsigned short* o = Bp + (size_t)t * 16;
#pragma unroll
    for (int i = 0; i < 16; ++i) {
        int k = kt * 32 + h * 16 + i;
        o[i] = f32_to_bf16(W[(size_t)k * Ncols + n]);
    }
}

// ---------------- WMMA GEMM: D[N, NT*16] = A[N,128] x B[128, NT*16] ----------------
// One block per 16-row tile; wave w computes columns [16w, 16w+16).
template <int NT>
__global__ void gemm_wmma_kernel(const unsigned short* __restrict__ Ap,
                                 const unsigned short* __restrict__ Bp,
                                 float* __restrict__ Dout) {
    const int D = NT * 16;
    int wave = threadIdx.x >> 5;       // 0..NT-1
    int lane = threadIdx.x & 31;
    int rt = blockIdx.x;

    const unsigned short* aBase = Ap + ((size_t)rt * 4) * 512 + (size_t)lane * 16;
    const unsigned short* bBase = Bp + (size_t)wave * 512 + (size_t)lane * 16;

    v8f c = {};
#pragma unroll
    for (int kt = 0; kt < 4; ++kt) {
        v16bf a = *(const v16bf*)(aBase + (size_t)kt * 512);
        v16bf b = *(const v16bf*)(bBase + (size_t)kt * NT * 512);
        c = __builtin_amdgcn_wmma_f32_16x16x32_bf16(false, a, false, b,
                                                    (short)0, c, false, false);
    }

    int h = lane >> 4;
    int n = wave * 16 + (lane & 15);
#pragma unroll
    for (int r = 0; r < 8; ++r) {
        Dout[(size_t)(rt * 16 + r + 8 * h) * D + n] = c[r];
    }
}

// ---------------- edge scatter: agg[dst] += h[src] * dinv[src]*dinv[dst] ----------------
template <int D>
__global__ void scatter_kernel(const float* __restrict__ h,
                               const long long* __restrict__ src,
                               const long long* __restrict__ dst,
                               const float* __restrict__ dinv,
                               float* __restrict__ agg, int E) {
    const int TPE = D / 4;  // threads per edge (float4 each)
    long long t = (long long)blockIdx.x * blockDim.x + threadIdx.x;
    if (t >= (long long)E * TPE) return;
    int e = (int)(t / TPE);
    int c = (int)(t % TPE) * 4;
    int s = (int)src[e];
    int d = (int)dst[e];
    float coef = dinv[s] * dinv[d];
    const float4 v = *(const float4*)(h + (size_t)s * D + c);
    float* a = agg + (size_t)d * D + c;
    atomicAdd(a + 0, v.x * coef);
    atomicAdd(a + 1, v.y * coef);
    atomicAdd(a + 2, v.z * coef);
    atomicAdd(a + 3, v.w * coef);
}

// ---------------- finalize: agg = [relu](agg + h * dinv^2 + bias) ----------------
template <int D, bool RELU>
__global__ void finalize_kernel(float* __restrict__ agg, const float* __restrict__ h,
                                const float* __restrict__ dinv, const float* __restrict__ bias,
                                int n) {
    int t = blockIdx.x * blockDim.x + threadIdx.x;
    if (t >= n * D) return;
    int i = t / D;
    int c = t - i * D;
    float di = dinv[i];
    float v = agg[t] + h[t] * di * di + bias[c];
    agg[t] = RELU ? fmaxf(v, 0.0f) : v;
}

extern "C" void kernel_launch(void* const* d_in, const int* in_sizes, int n_in,
                              void* d_out, int out_size, void* d_ws, size_t ws_size,
                              hipStream_t stream) {
    (void)in_sizes; (void)n_in; (void)out_size; (void)ws_size;

    constexpr int N = 50000, E = 800000, IN = 128, HID = 128, OUTC = 64;
    constexpr int RT = N / 16;  // 3125 row tiles (exact)

    const float*     x   = (const float*)d_in[0];
    const long long* ei  = (const long long*)d_in[1];   // int64 [2, E]
    const float*     W1  = (const float*)d_in[2];
    const float*     b1  = (const float*)d_in[3];
    const float*     W2  = (const float*)d_in[4];
    const float*     b2  = (const float*)d_in[5];
    float*           out = (float*)d_out;

    const long long* srcIdx = ei;
    const long long* dstIdx = ei + E;

    // ---- workspace carve-out (256B aligned) ----
    char* ws = (char*)d_ws;
    size_t off = 0;
    auto alloc = [&](size_t bytes) -> void* {
        void* p = ws + off;
        off = (off + bytes + 255) & ~(size_t)255;
        return p;
    };
    float* deg   = (float*)alloc((size_t)N * 4);
    float* dinv  = (float*)alloc((size_t)N * 4);
    float* h1    = (float*)alloc((size_t)N * HID * 4);   // raw GEMM1 output
    float* agg1  = (float*)alloc((size_t)N * HID * 4);   // scatter accum -> relu(h) in place
    float* h2    = (float*)alloc((size_t)N * OUTC * 4);  // raw GEMM2 output
    unsigned short* xA  = (unsigned short*)alloc((size_t)N * IN * 2);
    unsigned short* hA  = (unsigned short*)alloc((size_t)N * HID * 2);
    unsigned short* Wb1 = (unsigned short*)alloc((size_t)IN * HID * 2);
    unsigned short* Wb2 = (unsigned short*)alloc((size_t)HID * OUTC * 2);

    const int TB = 256;

    // 1) zero accumulators
    zero_kernel<<<2048, TB, 0, stream>>>(deg, N);
    zero_kernel<<<4096, TB, 0, stream>>>(agg1, N * HID);
    zero_kernel<<<4096, TB, 0, stream>>>(out, N * OUTC);

    // 2) pack weights and features into WMMA bf16 layouts
    pack_b_kernel<<<(4 * 8 * 32 + TB - 1) / TB, TB, 0, stream>>>(W1, Wb1, 8);
    pack_b_kernel<<<(4 * 4 * 32 + TB - 1) / TB, TB, 0, stream>>>(W2, Wb2, 4);
    pack_a_kernel<<<(RT * 128 + TB - 1) / TB, TB, 0, stream>>>(x, xA, RT);

    // 3) degree + normalization
    deg_kernel<<<(E + TB - 1) / TB, TB, 0, stream>>>(dstIdx, deg, E);
    dinv_kernel<<<(N + TB - 1) / TB, TB, 0, stream>>>(deg, dinv, N);

    // 4) layer 1: h1 = x @ W1 (WMMA, 8 waves per block = 16x128 tile)
    gemm_wmma_kernel<8><<<RT, 8 * 32, 0, stream>>>(xA, Wb1, h1);

    // 5) layer 1 aggregation
    {
        long long tot = (long long)E * (HID / 4);
        scatter_kernel<HID><<<(int)((tot + TB - 1) / TB), TB, 0, stream>>>(h1, srcIdx, dstIdx, dinv, agg1, E);
    }
    finalize_kernel<HID, true><<<(N * HID + TB - 1) / TB, TB, 0, stream>>>(agg1, h1, dinv, b1, N);

    // 6) repack relu output for layer 2 GEMM
    pack_a_kernel<<<(RT * 128 + TB - 1) / TB, TB, 0, stream>>>(agg1, hA, RT);

    // 7) layer 2: h2 = relu_h @ W2 (WMMA, 4 waves per block = 16x64 tile)
    gemm_wmma_kernel<4><<<RT, 4 * 32, 0, stream>>>(hA, Wb2, h2);

    // 8) layer 2 aggregation straight into d_out
    {
        long long tot = (long long)E * (OUTC / 4);
        scatter_kernel<OUTC><<<(int)((tot + TB - 1) / TB), TB, 0, stream>>>(h2, srcIdx, dstIdx, dinv, out, E);
    }
    finalize_kernel<OUTC, false><<<(N * OUTC + TB - 1) / TB, TB, 0, stream>>>(out, h2, dinv, b2, N);
}
